// GraphSAGEMemory_18674517803656
// MI455X (gfx1250) — compile-verified, matrix operand order checked by reading, not run on hardware
//
#include <hip/hip_runtime.h>

#define NN   8192
#define EE   262144
#define DIN  386
#define DINP 416   // 13 * 32
#define HH   128
#define NHD  4
#define HD   32

typedef __bf16 bf16x16 __attribute__((ext_vector_type(16)));
typedef float  v8f     __attribute__((ext_vector_type(8)));
typedef unsigned u32x4 __attribute__((ext_vector_type(4)));
typedef int      i32x4 __attribute__((ext_vector_type(4)));
typedef int      i32x8 __attribute__((ext_vector_type(8)));

struct Frag { union { bf16x16 v; unsigned u[8]; }; };

// Build a 16x32 bf16 fragment (A-matrix layout; also B when source is [n][k]).
// Lane L: row = L&15 ; K offset +8 for lanes 16..31 ; VGPR r holds K pair
// (2r, 2r+1) for r<4 and (16 + 2(r-4), +1) for r>=4.
__device__ inline Frag load_frag(const __bf16* base, int stride, int lane, int k0) {
  Frag f;
  const __bf16* p = base + (size_t)(lane & 15) * stride + k0 + ((lane & 16) >> 1);
  #pragma unroll
  for (int r = 0; r < 8; ++r) {
    int k = (r < 4) ? (2 * r) : (8 + 2 * r);
    f.u[r] = *reinterpret_cast<const unsigned*>(p + k);  // 2 packed bf16
  }
  return f;
}

__device__ inline v8f wmma_bf16(const Frag& a, const Frag& b, v8f c) {
  return __builtin_amdgcn_wmma_f32_16x16x32_bf16(false, a.v, false, b.v,
                                                 (short)0, c, false, false);
}

// ---- Tensor Data Mover: 2-D tile load Global -> LDS (D# per CDNA5 ISA §8) ---
// data_size = 4 bytes; tile = nrows x row_dwords, global row stride in dwords.
// 1-D contiguous loads are the nrows==1 case. 6-arg builtin (clang-23 form).
__device__ inline void tdm_load(const void* gsrc, unsigned lds_byte_off,
                                int row_dwords, int nrows, long long stride_dwords) {
  unsigned long long ga = (unsigned long long)(size_t)gsrc;
  u32x4 g0;
  g0[0] = 1u;                                            // count=1 (valid), user mode
  g0[1] = lds_byte_off;                                  // lds_addr
  g0[2] = (unsigned)(ga & 0xFFFFFFFFu);                  // global_addr[31:0]
  g0[3] = (unsigned)((ga >> 32) & 0x01FFFFFFu) | (2u << 30);  // addr[56:32] | type=2
  unsigned td0 = (unsigned)row_dwords;                   // tensor_dim0 (== tile: no OOB)
  unsigned td1 = (unsigned)nrows;
  i32x8 g1;
  g1[0] = (int)(2u << 16);                               // data_size = 4B, no flags
  g1[1] = (int)((td0 & 0xFFFFu) << 16);                  // tensor_dim0[15:0] @ 63:48
  g1[2] = (int)(((td0 >> 16) & 0xFFFFu) | ((td1 & 0xFFFFu) << 16)); // td0 hi | td1 lo
  g1[3] = (int)(((td1 >> 16) & 0xFFFFu) | ((td0 & 0xFFFFu) << 16)); // td1 hi | tile_dim0
  g1[4] = (int)(td1 & 0xFFFFu);                          // tile_dim1 (tile_dim2 = 0)
  g1[5] = (int)((unsigned long long)stride_dwords & 0xFFFFFFFFu);   // dim0_stride lo
  g1[6] = (int)(((unsigned long long)stride_dwords >> 32) & 0xFFFFu); // dim0_stride hi
  g1[7] = 0;
  i32x4 z4 = {0, 0, 0, 0};
  i32x8 z8 = {0, 0, 0, 0, 0, 0, 0, 0};
  __builtin_amdgcn_tensor_load_to_lds(g0, g1, z4, z4, z8, 0);
}

// ---------------------------------------------------------------- utility ---
__global__ __launch_bounds__(256) void k_zero(float* p, int n) {
  int i = blockIdx.x * 256 + threadIdx.x;
  if (i < n) p[i] = 0.0f;
}

__global__ __launch_bounds__(256) void k_count(const int* dst, float* cnt, int e) {
  int i = blockIdx.x * 256 + threadIdx.x;
  if (i < e)
    __hip_atomic_fetch_add(&cnt[dst[i]], 1.0f, __ATOMIC_RELAXED, __HIP_MEMORY_SCOPE_AGENT);
}

__global__ __launch_bounds__(256) void k_scatter(const float* __restrict__ x,
                                                 const int* __restrict__ src,
                                                 const int* __restrict__ dst,
                                                 float* agg, int e, int cols, int chunks) {
  int gid = blockIdx.x * 256 + threadIdx.x;
  int total = e * chunks;
  if (gid >= total) return;
  int ed = gid / chunks;
  int f0 = (gid - ed * chunks) * 4;
  int s = src[ed], d = dst[ed];
  const float* xs = x + (size_t)s * cols;
  float* ad = agg + (size_t)d * cols;
  int fend = f0 + 4; if (fend > cols) fend = cols;
  for (int f = f0; f < fend; ++f)
    __hip_atomic_fetch_add(&ad[f], xs[f], __ATOMIC_RELAXED, __HIP_MEMORY_SCOPE_AGENT);
}

// f32[rows,sc] -> bf16[rows,dc] (zero-padded); optional per-row mean divide
__global__ __launch_bounds__(256) void k_cvt(__bf16* dst, const float* __restrict__ s,
                                             int rows, int sc, int dc,
                                             const float* __restrict__ cnt) {
  int gid = blockIdx.x * 256 + threadIdx.x;
  if (gid >= rows * dc) return;
  int r = gid / dc, c = gid - r * dc;
  float v = 0.0f;
  if (c < sc) {
    v = s[(size_t)r * sc + c];
    if (cnt) v /= fmaxf(cnt[r], 1.0f);
  }
  dst[gid] = (__bf16)v;
}

// ------------------------------------------------------------- WMMA GEMM ----
// out[m, n0:] = act( A1[m,:K1] @ W1[n,:K1]^T + bias + A2 @ W2^T ) (+ residual)
// grid.x = M/16 tile, (grid.y*8 + wave) = N/16 tile. 256 threads = 8 waves.
// A tiles staged via TDM (contiguous 16 x K block).
template <int K1, int K2>
__global__ __launch_bounds__(256) void k_gemm(const __bf16* __restrict__ A1,
                                              const __bf16* __restrict__ W1,
                                              const __bf16* __restrict__ A2,
                                              const __bf16* __restrict__ W2,
                                              const float* __restrict__ bias,
                                              const float* __restrict__ residual,
                                              float* __restrict__ outF,
                                              __bf16* __restrict__ outB,
                                              int oc, int relu) {
  __shared__ __attribute__((aligned(16))) __bf16 lds[16 * K1 + 16 * (K2 > 0 ? K2 : 1)];
  int lane = threadIdx.x & 31, wave = threadIdx.x >> 5;
  int m0 = blockIdx.x * 16;
  int n0 = (blockIdx.y * 8 + wave) * 16;

  if (threadIdx.x < 32) {  // one wave drives the Tensor Data Mover
    tdm_load(A1 + (size_t)m0 * K1, (unsigned)(size_t)&lds[0], 16 * K1 / 2, 1, 16 * K1 / 2);
    if constexpr (K2 > 0)
      tdm_load(A2 + (size_t)m0 * K2, (unsigned)(size_t)&lds[16 * K1],
               16 * K2 / 2, 1, 16 * K2 / 2);
    __builtin_amdgcn_s_wait_tensorcnt(0);
  }
  __syncthreads();

  v8f acc = {};
  const __bf16* wrow1 = W1 + (size_t)n0 * K1;
  #pragma unroll
  for (int k0 = 0; k0 < K1; k0 += 32) {
    Frag a = load_frag(lds, K1, lane, k0);
    Frag b = load_frag(wrow1, K1, lane, k0);
    acc = wmma_bf16(a, b, acc);
  }
  if constexpr (K2 > 0) {
    const __bf16* wrow2 = W2 + (size_t)n0 * K2;
    const __bf16* l2 = lds + 16 * K1;
    #pragma unroll
    for (int k0 = 0; k0 < K2; k0 += 32) {
      Frag a = load_frag(l2, K2, lane, k0);
      Frag b = load_frag(wrow2, K2, lane, k0);
      acc = wmma_bf16(a, b, acc);
    }
  }

  int col = lane & 15;
  int rbase = (lane & 16) >> 1;           // +8 rows for upper half-wave
  float bv = bias ? bias[n0 + col] : 0.0f;
  #pragma unroll
  for (int r = 0; r < 8; ++r) {
    int row = m0 + r + rbase;
    float v = acc[r] + bv;
    if (relu) v = fmaxf(v, 0.0f);
    if (residual) v += residual[(size_t)row * oc + n0 + col];
    size_t idx = (size_t)row * oc + n0 + col;
    outF[idx] = v;
    if (outB) outB[idx] = (__bf16)v;
  }
}

// ------------------------------------------------------------------- GRU ----
__global__ __launch_bounds__(256) void k_gru(const float* __restrict__ gi,
                                             const float* __restrict__ gh,
                                             const float* __restrict__ mold,
                                             float* __restrict__ memF,
                                             __bf16* __restrict__ memB) {
  int gid = blockIdx.x * 256 + threadIdx.x;
  if (gid >= NN * HH) return;
  int n = gid >> 7, j = gid & 127;
  const float* gin = gi + (size_t)n * 384;
  const float* ghn = gh + (size_t)n * 384;
  float r = 1.0f / (1.0f + __expf(-(gin[j] + ghn[j])));
  float z = 1.0f / (1.0f + __expf(-(gin[128 + j] + ghn[128 + j])));
  float cand = tanhf(gin[256 + j] + r * ghn[256 + j]);
  float m = (1.0f - z) * cand + z * mold[gid];
  memF[gid] = m;
  memB[gid] = (__bf16)m;
}

// qkv [N,384] f32 -> per-head Q (pre-scaled) / K row-major, V transposed [h][d][n]
__global__ __launch_bounds__(256) void k_qkv_split(const float* __restrict__ qkv,
                                                   __bf16* __restrict__ Qh,
                                                   __bf16* __restrict__ Kh,
                                                   __bf16* __restrict__ Vt) {
  int gid = blockIdx.x * 256 + threadIdx.x;
  if (gid >= NN * HH) return;
  int n = gid >> 7, j = gid & 127;
  int h = j >> 5, d = j & 31;
  const float* row = qkv + (size_t)n * 384;
  Qh[((size_t)h * NN + n) * HD + d] = (__bf16)(row[j] * 0.1767766953f); // 1/sqrt(32)
  Kh[((size_t)h * NN + n) * HD + d] = (__bf16)(row[128 + j]);
  Vt[(size_t)h * HD * NN + (size_t)d * NN + n] = (__bf16)(row[256 + j]);
}

// ------------------------------------------------- flash attention (WMMA) ---
// grid: (NN/128, NHD), 8 waves/block, each wave owns a 16-query tile.
// Transposed formulation: S^T = K_tile @ Q^T so queries sit in lanes -> the
// softmax max/sum reductions are per-lane + ONE shfl_xor(16); O^T = V^T @ P^T.
// K/V tiles (128 keys) staged in LDS once per block via the TDM (V is a true
// strided 2-D tile) and shared by all 8 waves.
__global__ __launch_bounds__(256) void k_attn(const __bf16* __restrict__ Qh,
                                              const __bf16* __restrict__ Kh,
                                              const __bf16* __restrict__ Vt,
                                              __bf16* __restrict__ attB) {
  __shared__ __attribute__((aligned(16))) __bf16 ldsK[128 * HD];   // [key][d]
  __shared__ __attribute__((aligned(16))) __bf16 ldsV[HD * 128];   // [d][key]
  __shared__ __attribute__((aligned(16))) __bf16 ldsP[8][16 * 32]; // [query][key]
  int lane = threadIdx.x & 31, wave = threadIdx.x >> 5;
  int head = blockIdx.y;
  int q0 = (blockIdx.x * 8 + wave) * 16;
  const __bf16* Q = Qh + (size_t)head * NN * HD;
  const __bf16* K = Kh + (size_t)head * NN * HD;
  const __bf16* V = Vt + (size_t)head * HD * NN;

  Frag bq = load_frag(Q + (size_t)q0 * HD, HD, lane, 0);  // B: [d x queries]
  v8f o0 = {}, o1 = {};
  float m = -3.0e38f, l = 0.0f;                           // per-lane (query=lane&15)
  int col = lane & 15;
  int rbase = (lane & 16) >> 1;
  __bf16* pbuf = &ldsP[wave][0];

  for (int s0 = 0; s0 < NN; s0 += 128) {
    if (threadIdx.x < 32) {
      tdm_load(K + (size_t)s0 * HD, (unsigned)(size_t)&ldsK[0], 128 * HD / 2, 1,
               128 * HD / 2);                                  // contiguous keys
      tdm_load(V + s0, (unsigned)(size_t)&ldsV[0], 128 / 2, HD, NN / 2); // strided
      __builtin_amdgcn_s_wait_tensorcnt(0);
    }
    __syncthreads();

    #pragma unroll
    for (int t = 0; t < 128; t += 32) {
      Frag ak0 = load_frag(ldsK + (size_t)t * HD, HD, lane, 0);        // keys t..t+15
      Frag ak1 = load_frag(ldsK + (size_t)(t + 16) * HD, HD, lane, 0); // keys t+16..t+31
      v8f z = {};
      v8f st0 = wmma_bf16(ak0, bq, z);   // S^T [key r+rbase][query col]
      v8f st1 = wmma_bf16(ak1, bq, z);

      float mx = fmaxf(st0[0], st1[0]);
      #pragma unroll
      for (int r = 1; r < 8; ++r) mx = fmaxf(mx, fmaxf(st0[r], st1[r]));
      mx = fmaxf(mx, __shfl_xor(mx, 16));
      float mnew = fmaxf(m, mx);
      float corr = __expf(m - mnew);
      float rs = 0.0f;
      #pragma unroll
      for (int r = 0; r < 8; ++r) {
        float p0 = __expf(st0[r] - mnew);
        float p1 = __expf(st1[r] - mnew);
        rs += p0 + p1;
        pbuf[col * 32 + r + rbase]      = (__bf16)p0;  // P^T -> [query][key] in LDS
        pbuf[col * 32 + 16 + r + rbase] = (__bf16)p1;
      }
      rs += __shfl_xor(rs, 16);
      l = l * corr + rs;
      m = mnew;
      #pragma unroll
      for (int r = 0; r < 8; ++r) { o0[r] *= corr; o1[r] *= corr; }

      Frag bp  = load_frag(pbuf, 32, lane, 0);                  // B: [key x query]
      Frag av0 = load_frag(ldsV + t, 128, lane, 0);             // A: d = 0..15
      Frag av1 = load_frag(ldsV + 16 * 128 + t, 128, lane, 0);  // A: d = 16..31
      o0 = wmma_bf16(av0, bp, o0);   // O^T [d][query]
      o1 = wmma_bf16(av1, bp, o1);
    }
    __syncthreads();
  }

  float inv = 1.0f / l;
  #pragma unroll
  for (int r = 0; r < 8; ++r) {
    int d = r + rbase;
    attB[(size_t)(q0 + col) * HH + head * HD + d]      = (__bf16)(o0[r] * inv);
    attB[(size_t)(q0 + col) * HH + head * HD + 16 + d] = (__bf16)(o1[r] * inv);
  }
}

// ---------------------------------------------------------------- tails -----
__global__ __launch_bounds__(256) void k_combine(const float* __restrict__ h,
                                                 const float* __restrict__ atto,
                                                 float* __restrict__ out) {
  int gid = blockIdx.x * 256 + threadIdx.x;
  if (gid < NN * HH) out[gid] = h[gid] + atto[gid];
}

__global__ __launch_bounds__(256) void k_heads(const float* __restrict__ comb,
                                               const float* __restrict__ W_act,
                                               const float* __restrict__ b_act,
                                               const float* __restrict__ W_emo,
                                               const float* __restrict__ b_emo,
                                               float* __restrict__ actO,
                                               float* __restrict__ emoO) {
  int n = blockIdx.x * 256 + threadIdx.x;
  if (n >= NN) return;
  const float* c = comb + (size_t)n * HH;
  float a = b_act[0], e0 = b_emo[0], e1 = b_emo[1];
  for (int j = 0; j < HH; ++j) {
    float v = c[j];
    a  += v * W_act[j];
    e0 += v * W_emo[j];
    e1 += v * W_emo[HH + j];
  }
  actO[n] = 1.0f / (1.0f + __expf(-a));
  emoO[2 * n]     = tanhf(e0);
  emoO[2 * n + 1] = tanhf(e1);
}

// ================================================================= launch ===
extern "C" void kernel_launch(void* const* d_in, const int* in_sizes, int n_in,
                              void* d_out, int out_size, void* d_ws, size_t ws_size,
                              hipStream_t stream) {
  (void)in_sizes; (void)n_in; (void)out_size; (void)ws_size;

  const float* x    = (const float*)d_in[0];
  const int*   ei   = (const int*)d_in[1];
  const float* mems = (const float*)d_in[2];
  const float* Wl0 = (const float*)d_in[3];  const float* bl0 = (const float*)d_in[4];
  const float* Wr0 = (const float*)d_in[5];
  const float* Wl1 = (const float*)d_in[6];  const float* bl1 = (const float*)d_in[7];
  const float* Wr1 = (const float*)d_in[8];
  const float* Wl2 = (const float*)d_in[9];  const float* bl2 = (const float*)d_in[10];
  const float* Wr2 = (const float*)d_in[11];
  const float* W_ih = (const float*)d_in[12]; const float* W_hh = (const float*)d_in[13];
  const float* b_ih = (const float*)d_in[14]; const float* b_hh = (const float*)d_in[15];
  const float* W_qkv = (const float*)d_in[16]; const float* b_qkv = (const float*)d_in[17];
  const float* W_o = (const float*)d_in[18];   const float* b_o = (const float*)d_in[19];
  const float* W_act = (const float*)d_in[20]; const float* b_act = (const float*)d_in[21];
  const float* W_emo = (const float*)d_in[22]; const float* b_emo = (const float*)d_in[23];

  const int* src = ei;
  const int* dst = ei + EE;

  float* out = (float*)d_out;
  float* outComb = out;
  float* outAct  = out + (size_t)NN * HH;
  float* outEmo  = outAct + NN;
  float* outMem  = outEmo + 2 * NN;

  // ---- workspace bump allocator
  char* wsp = (char*)d_ws;
  size_t off = 0;
  auto alloc = [&](size_t bytes) -> void* {
    void* p = wsp + off;
    off += (bytes + 255) & ~(size_t)255;
    return p;
  };
  float*  cnt  = (float*)alloc((size_t)NN * 4);
  float*  bufA = (float*)alloc((size_t)NN * DIN * 4);  // agg0 / agg128 / gi / qkv
  float*  bufB = (float*)alloc((size_t)NN * 384 * 4);  // gh / atto
  __bf16* xb   = (__bf16*)alloc((size_t)NN * DINP * 2);
  __bf16* aggb = (__bf16*)alloc((size_t)NN * DINP * 2);
  float*  h0   = (float*)alloc((size_t)NN * HH * 4);
  float*  h1   = (float*)alloc((size_t)NN * HH * 4);
  __bf16* hb0  = (__bf16*)alloc((size_t)NN * HH * 2);
  __bf16* hb1  = (__bf16*)alloc((size_t)NN * HH * 2);
  __bf16* memb = (__bf16*)alloc((size_t)NN * HH * 2);
  __bf16* Qh   = (__bf16*)alloc((size_t)NN * HH * 2);
  __bf16* Kh   = (__bf16*)alloc((size_t)NN * HH * 2);
  __bf16* Vt   = (__bf16*)alloc((size_t)NN * HH * 2);
  __bf16* attb = (__bf16*)alloc((size_t)NN * HH * 2);
  __bf16* wl0b = (__bf16*)alloc((size_t)HH * DINP * 2);
  __bf16* wr0b = (__bf16*)alloc((size_t)HH * DINP * 2);
  __bf16* wl1b = (__bf16*)alloc((size_t)HH * HH * 2);
  __bf16* wr1b = (__bf16*)alloc((size_t)HH * HH * 2);
  __bf16* wl2b = (__bf16*)alloc((size_t)HH * HH * 2);
  __bf16* wr2b = (__bf16*)alloc((size_t)HH * HH * 2);
  __bf16* wihb = (__bf16*)alloc((size_t)384 * HH * 2);
  __bf16* whhb = (__bf16*)alloc((size_t)384 * HH * 2);
  __bf16* wqkb = (__bf16*)alloc((size_t)384 * HH * 2);
  __bf16* wob  = (__bf16*)alloc((size_t)HH * HH * 2);

  auto blocks = [](long long n) { return (unsigned)((n + 255) / 256); };

  // ---- weight conversions to bf16 (K padded to WMMA multiple)
  k_cvt<<<blocks((long long)HH * DINP), 256, 0, stream>>>(wl0b, Wl0, HH, DIN, DINP, nullptr);
  k_cvt<<<blocks((long long)HH * DINP), 256, 0, stream>>>(wr0b, Wr0, HH, DIN, DINP, nullptr);
  k_cvt<<<blocks((long long)HH * HH), 256, 0, stream>>>(wl1b, Wl1, HH, HH, HH, nullptr);
  k_cvt<<<blocks((long long)HH * HH), 256, 0, stream>>>(wr1b, Wr1, HH, HH, HH, nullptr);
  k_cvt<<<blocks((long long)HH * HH), 256, 0, stream>>>(wl2b, Wl2, HH, HH, HH, nullptr);
  k_cvt<<<blocks((long long)HH * HH), 256, 0, stream>>>(wr2b, Wr2, HH, HH, HH, nullptr);
  k_cvt<<<blocks(384LL * HH), 256, 0, stream>>>(wihb, W_ih, 384, HH, HH, nullptr);
  k_cvt<<<blocks(384LL * HH), 256, 0, stream>>>(whhb, W_hh, 384, HH, HH, nullptr);
  k_cvt<<<blocks(384LL * HH), 256, 0, stream>>>(wqkb, W_qkv, 384, HH, HH, nullptr);
  k_cvt<<<blocks((long long)HH * HH), 256, 0, stream>>>(wob, W_o, HH, HH, HH, nullptr);

  // ---- degree counts (shared by all SAGE layers)
  k_zero<<<blocks(NN), 256, 0, stream>>>(cnt, NN);
  k_count<<<blocks(EE), 256, 0, stream>>>(dst, cnt, EE);

  // ---- SAGE layer 0: h0 = relu(mean_agg(x) @ Wl0^T + bl0 + x @ Wr0^T)
  k_zero<<<blocks((long long)NN * DIN), 256, 0, stream>>>(bufA, NN * DIN);
  k_scatter<<<blocks((long long)EE * 97), 256, 0, stream>>>(x, src, dst, bufA, EE, DIN, 97);
  k_cvt<<<blocks((long long)NN * DINP), 256, 0, stream>>>(aggb, bufA, NN, DIN, DINP, cnt);
  k_cvt<<<blocks((long long)NN * DINP), 256, 0, stream>>>(xb, x, NN, DIN, DINP, nullptr);
  k_gemm<DINP, DINP><<<dim3(NN / 16, 1), 256, 0, stream>>>(
      aggb, wl0b, xb, wr0b, bl0, nullptr, h0, hb0, HH, 1);

  // ---- SAGE layer 1: h1 = h0 + relu(mean_agg(h0) @ Wl1^T + bl1 + h0 @ Wr1^T)
  k_zero<<<blocks((long long)NN * HH), 256, 0, stream>>>(bufA, NN * HH);
  k_scatter<<<blocks((long long)EE * 32), 256, 0, stream>>>(h0, src, dst, bufA, EE, HH, 32);
  k_cvt<<<blocks((long long)NN * HH), 256, 0, stream>>>(aggb, bufA, NN, HH, HH, cnt);
  k_gemm<HH, HH><<<dim3(NN / 16, 1), 256, 0, stream>>>(
      aggb, wl1b, hb0, wr1b, bl1, h0, h1, hb1, HH, 1);

  // ---- SAGE layer 2: h0 = h1 + relu(...)
  k_zero<<<blocks((long long)NN * HH), 256, 0, stream>>>(bufA, NN * HH);
  k_scatter<<<blocks((long long)EE * 32), 256, 0, stream>>>(h1, src, dst, bufA, EE, HH, 32);
  k_cvt<<<blocks((long long)NN * HH), 256, 0, stream>>>(aggb, bufA, NN, HH, HH, cnt);
  k_gemm<HH, HH><<<dim3(NN / 16, 1), 256, 0, stream>>>(
      aggb, wl2b, hb1, wr2b, bl2, h1, h0, hb0, HH, 1);
  // final h = h0 (f32), hb0 (bf16)

  // ---- GRU: gi = h @ W_ih^T + b_ih ; gh = mem @ W_hh^T + b_hh
  k_gemm<HH, 0><<<dim3(NN / 16, 3), 256, 0, stream>>>(
      hb0, wihb, nullptr, nullptr, b_ih, nullptr, bufA, nullptr, 384, 0);
  k_cvt<<<blocks((long long)NN * HH), 256, 0, stream>>>(memb, mems, NN, HH, HH, nullptr);
  k_gemm<HH, 0><<<dim3(NN / 16, 3), 256, 0, stream>>>(
      memb, whhb, nullptr, nullptr, b_hh, nullptr, bufB, nullptr, 384, 0);
  k_gru<<<blocks((long long)NN * HH), 256, 0, stream>>>(bufA, bufB, mems, outMem, memb);

  // ---- QKV projection + per-head split (Q pre-scaled, V transposed)
  k_gemm<HH, 0><<<dim3(NN / 16, 3), 256, 0, stream>>>(
      memb, wqkb, nullptr, nullptr, b_qkv, nullptr, bufA, nullptr, 384, 0);
  k_qkv_split<<<blocks((long long)NN * HH), 256, 0, stream>>>(bufA, Qh, Kh, Vt);

  // ---- flash multi-head self-attention (WMMA + TDM-staged K/V)
  k_attn<<<dim3(NN / 128, NHD), 256, 0, stream>>>(Qh, Kh, Vt, attb);

  // ---- output projection + combine + heads
  k_gemm<HH, 0><<<dim3(NN / 16, 1), 256, 0, stream>>>(
      attb, wob, nullptr, nullptr, b_o, nullptr, bufB, nullptr, HH, 0);
  k_combine<<<blocks((long long)NN * HH), 256, 0, stream>>>(h0, bufB, outComb);
  k_heads<<<blocks(NN), 256, 0, stream>>>(outComb, W_act, b_act, W_emo, b_emo, outAct, outEmo);
}
